// Mamba2Block_61924838474173
// MI455X (gfx1250) — compile-verified, matrix operand order checked by reading, not run on hardware
//
#include <hip/hip_runtime.h>
#include <hip/hip_bf16.h>
#include <math.h>

// ---------------------------------------------------------------------------
// Mamba2-style block for MI455X (gfx1250): bf16 WMMA everywhere.
//   BATCH=16, SEQ=2048, DM=1024
// Pipeline:
//   1) convert weights fp32->bf16 (A,B,C,W1,W2), init state slot0 + barrier ctr
//   2) U = X @ B^T                      (WMMA GEMM, fp32 out)
//   3) scan: s_t = tanh(s_{t-1}A^T+u)   (16 WGs; A-slice staged into 128KB LDS
//                                        via global_load_async_to_lds_b128;
//                                        4-way K-split WMMA chains;
//                                        global atomic barrier per step)
//   4) fused: Y = S@C^T + D*X (in LDS) -> H=silu(Y@W1^T+b1) (in LDS)
//             -> out = Y * sigmoid(H@W2^T+b2)        (Y never hits HBM)
// ---------------------------------------------------------------------------

#define BATCH  16
#define SEQ    2048
#define DM     1024
#define DM2    2048
#define NBLK   16   // scan workgroups

typedef __bf16 bf16;
typedef __attribute__((ext_vector_type(16))) __bf16 v16bf;
typedef __attribute__((ext_vector_type(8)))  __bf16 v8bf;
typedef __attribute__((ext_vector_type(8)))  float  v8f;

// ---- WMMA fragment helpers (CDNA5 16x16x32 bf16 layouts, ISA 7.12.2) ------
// A-frag: lane l -> M = l&15 ; K halves: kh=(l>>4)*8 -> elems kh+[0..7], kh+16+[0..7]
// B-frag (transposed weight, W row-major [N,K]): lane l -> N = l&15, same K pattern.
__device__ __forceinline__ v16bf frag_bf16(const bf16* p) {
  v8bf lo = *(const v8bf*)(p);
  v8bf hi = *(const v8bf*)(p + 16);
  v16bf r;
#pragma unroll
  for (int i = 0; i < 8; ++i) { r[i] = lo[i]; r[i + 8] = hi[i]; }
  return r;
}
__device__ __forceinline__ v16bf frag_f32(const float* p) {
  v16bf r;
#pragma unroll
  for (int i = 0; i < 8; ++i) { r[i] = (bf16)p[i]; r[i + 8] = (bf16)p[16 + i]; }
  return r;
}
__device__ __forceinline__ v8f wmma_bf16(v16bf a, v16bf b, v8f c) {
  return __builtin_amdgcn_wmma_f32_16x16x32_bf16(false, a, false, b, (short)0, c,
                                                 false, false);
}
__device__ __forceinline__ float fast_tanh(float x) {
#if __has_builtin(__builtin_amdgcn_tanhf)
  return __builtin_amdgcn_tanhf(x);   // native v_tanh_f32 (TRANS)
#else
  return tanhf(x);
#endif
}
__device__ __forceinline__ void wait_asynccnt0() {
#if __has_builtin(__builtin_amdgcn_s_wait_asynccnt)
  __builtin_amdgcn_s_wait_asynccnt(0);
#else
  asm volatile("s_wait_asynccnt 0x0" ::: "memory");
#endif
}

// ---- small helpers --------------------------------------------------------
__global__ void k_cvt(bf16* __restrict__ dst, const float* __restrict__ src, int n) {
  int i = blockIdx.x * blockDim.x + threadIdx.x;
  if (i < n) dst[i] = (bf16)src[i];
}

__global__ void k_init(bf16* __restrict__ s0, const float* __restrict__ state,
                       unsigned* __restrict__ ctr) {
  int i = blockIdx.x * blockDim.x + threadIdx.x;
  if (i < BATCH * DM) s0[i] = (bf16)state[i];
  if (i == 0) *ctr = 0u;
}

// ---- generic GEMM: Out[M,N] = Af32[M,K] @ Wbf[N,K]^T ----------------------
// block = 128 threads (4 waves); wave -> one 16-row tile; block -> 64 cols.
__global__ void k_gemm_xw(const float* __restrict__ A, const bf16* __restrict__ W,
                          float* __restrict__ Out, int M, int N, int K) {
  const int lane = threadIdx.x & 31;
  const int wv   = threadIdx.x >> 5;
  const int m0   = (blockIdx.y * 4 + wv) * 16;
  const int n0   = blockIdx.x * 64;
  if (m0 >= M) return;
  const int half = lane >> 4, nn = lane & 15, kh = half * 8;
  const float* arow = A + (size_t)(m0 + (lane & 15)) * K;
  v8f acc[4];
#pragma unroll
  for (int j = 0; j < 4; ++j) acc[j] = (v8f){0, 0, 0, 0, 0, 0, 0, 0};
  for (int k = 0; k < K; k += 32) {
    v16bf a = frag_f32(arow + k + kh);
    __builtin_prefetch(arow + k + 128, 0, 1);
#pragma unroll
    for (int j = 0; j < 4; ++j) {
      const bf16* brow = W + (size_t)(n0 + j * 16 + nn) * K + k + kh;
      acc[j] = wmma_bf16(a, frag_bf16(brow), acc[j]);
    }
  }
#pragma unroll
  for (int j = 0; j < 4; ++j)
#pragma unroll
    for (int v = 0; v < 8; ++v)
      Out[(size_t)(m0 + v + 8 * half) * N + (n0 + j * 16 + nn)] = acc[j][v];
}

// ---- sequential scan: s_t = tanh(s_{t-1} @ A^T + u_t) ---------------------
// 16 blocks x 128 threads. Block b owns output cols [64b, 64b+64); its A-slice
// (64x1024 bf16 = 128KB) is staged in LDS via async global->LDS copies.
// States stored per-step in Sbuf[t] (bf16 [2049][16][1024]); device-wide
// atomic barrier separates steps. 4-way K-split breaks the WMMA RAW chain.
__global__ void k_scan(const bf16* __restrict__ Abf, const float* __restrict__ U,
                       bf16* __restrict__ Sbuf, float* __restrict__ outState,
                       unsigned* __restrict__ ctr) {
  extern __shared__ bf16 Alds[];  // 64*1024 bf16 = 128 KB
  const int tid  = threadIdx.x;
  const int lane = tid & 31;
  const int wv   = tid >> 5;      // 4 waves
  const int c0   = blockIdx.x * 64;

  {  // stage A-slice into LDS with ASYNCcnt-tracked async b128 copies
    const char*    gsrc = (const char*)(Abf + (size_t)c0 * DM);
    const unsigned lbase = (unsigned)(uintptr_t)(void*)Alds;
    for (int i = tid * 16; i < 64 * DM * 2; i += 128 * 16) {
      unsigned    laddr = lbase + (unsigned)i;
      const char* g     = gsrc + i;
      asm volatile("global_load_async_to_lds_b128 %0, %1, off"
                   :: "v"(laddr), "v"(g) : "memory");
    }
    wait_asynccnt0();
  }
  __syncthreads();

  const int half = lane >> 4, nn = lane & 15, kh = half * 8;
  const int nloc = wv * 16;  // wave's local 16-col tile
  const bf16* bbase = Alds + (size_t)(nloc + nn) * DM + kh;

  for (int t = 1; t <= SEQ; ++t) {
    const bf16* arow = Sbuf + (size_t)(t - 1) * (BATCH * DM) +
                       (size_t)(lane & 15) * DM + kh;
    v8f a0 = (v8f){0, 0, 0, 0, 0, 0, 0, 0};
    v8f a1 = a0, a2 = a0, a3 = a0;
    for (int k = 0; k < DM; k += 128) {  // 4 independent WMMA chains
      a0 = wmma_bf16(frag_bf16(arow + k),      frag_bf16(bbase + k),      a0);
      a1 = wmma_bf16(frag_bf16(arow + k + 32), frag_bf16(bbase + k + 32), a1);
      a2 = wmma_bf16(frag_bf16(arow + k + 64), frag_bf16(bbase + k + 64), a2);
      a3 = wmma_bf16(frag_bf16(arow + k + 96), frag_bf16(bbase + k + 96), a3);
    }
    v8f acc = (a0 + a1) + (a2 + a3);

#pragma unroll
    for (int v = 0; v < 8; ++v) {
      const int row = v + 8 * half;           // batch index
      const int col = c0 + nloc + nn;         // state dim
      float val = fast_tanh(acc[v] + U[((size_t)row * SEQ + (t - 1)) * DM + col]);
      Sbuf[(size_t)t * (BATCH * DM) + (size_t)row * DM + col] = (bf16)val;
      if (t == SEQ) outState[(size_t)row * DM + col] = val;
    }

    // device-wide step barrier (release stores, arrive, spin, acquire)
    __threadfence();
    __syncthreads();
    if (tid == 0) {
      atomicAdd(ctr, 1u);
      const unsigned target = (unsigned)NBLK * (unsigned)t;
      while (__hip_atomic_load(ctr, __ATOMIC_ACQUIRE, __HIP_MEMORY_SCOPE_AGENT) < target)
        __builtin_amdgcn_s_sleep(1);
    }
    __syncthreads();
    __threadfence();
  }
}

// ---- fused output path ----------------------------------------------------
// One block (256 thr, 8 waves) owns 16 rows (b, i0..i0+15).
//   phase 0: Yt[16][1024] (LDS, fp32) = S@C^T + D*X
//   phase 1: Ht[16][2048] (LDS, bf16) = silu(Yt@W1^T + b1)
//   phase 2: out = Yt * sigmoid(Ht@W2^T + b2)
// LDS: 64KB + 64KB = 128KB. Y never goes through HBM.
__global__ void k_fused(const bf16* __restrict__ Sbuf, const bf16* __restrict__ Cw,
                        const float* __restrict__ X, const float* __restrict__ Dv,
                        const bf16* __restrict__ W1, const float* __restrict__ b1,
                        const bf16* __restrict__ W2, const float* __restrict__ b2,
                        float* __restrict__ Out) {
  extern __shared__ char smem[];
  float* Yt = (float*)smem;                       // [16][1024] fp32
  bf16*  Ht = (bf16*)(smem + 16 * DM * 4);        // [16][2048] bf16

  const int tid  = threadIdx.x;
  const int lane = tid & 31;
  const int wv   = tid >> 5;                      // 8 waves
  const int half = lane >> 4, nn = lane & 15, kh = half * 8;

  const int    blk = blockIdx.x;                  // 0..2047
  const int    b   = blk >> 7;                    // batch
  const int    i0  = (blk & 127) * 16;            // seq tile
  const size_t m0  = (size_t)b * SEQ + i0;        // global row base

  // ---- phase 0: Y tile = S @ C^T + D*X  (row i uses state slot i+1)
  const bf16* arowS = Sbuf + (size_t)(i0 + (lane & 15) + 1) * (BATCH * DM) +
                      (size_t)b * DM + kh;
#pragma unroll
  for (int pass = 0; pass < 2; ++pass) {
    const int n0 = wv * 128 + pass * 64;
    v8f acc[4];
#pragma unroll
    for (int j = 0; j < 4; ++j) acc[j] = (v8f){0, 0, 0, 0, 0, 0, 0, 0};
    for (int k = 0; k < DM; k += 32) {
      v16bf a = frag_bf16(arowS + k);
#pragma unroll
      for (int j = 0; j < 4; ++j) {
        const bf16* brow = Cw + (size_t)(n0 + j * 16 + nn) * DM + k + kh;
        acc[j] = wmma_bf16(a, frag_bf16(brow), acc[j]);
      }
    }
#pragma unroll
    for (int j = 0; j < 4; ++j)
#pragma unroll
      for (int v = 0; v < 8; ++v) {
        const int row = v + 8 * half, c = n0 + j * 16 + nn;
        Yt[row * DM + c] = acc[j][v] + Dv[c] * X[(m0 + row) * DM + c];
      }
  }
  __syncthreads();

  // ---- phase 1: H = silu(Y @ W1^T + b1)
  const float* yrow = Yt + (lane & 15) * DM + kh;
#pragma unroll
  for (int pass = 0; pass < 4; ++pass) {
    const int n0 = wv * 256 + pass * 64;
    v8f acc[4];
#pragma unroll
    for (int j = 0; j < 4; ++j) acc[j] = (v8f){0, 0, 0, 0, 0, 0, 0, 0};
    for (int k = 0; k < DM; k += 32) {
      v16bf a = frag_f32(yrow + k);
#pragma unroll
      for (int j = 0; j < 4; ++j) {
        const bf16* brow = W1 + (size_t)(n0 + j * 16 + nn) * DM + k + kh;
        acc[j] = wmma_bf16(a, frag_bf16(brow), acc[j]);
      }
    }
#pragma unroll
    for (int j = 0; j < 4; ++j)
#pragma unroll
      for (int v = 0; v < 8; ++v) {
        const int row = v + 8 * half, col = n0 + j * 16 + nn;
        float h = acc[j][v] + b1[col];
        h = h / (1.0f + __expf(-h));  // SiLU
        Ht[(size_t)row * DM2 + col] = (bf16)h;
      }
  }
  __syncthreads();

  // ---- phase 2: out = Y * sigmoid(H @ W2^T + b2)
  const bf16* hrow = Ht + (size_t)(lane & 15) * DM2 + kh;
#pragma unroll
  for (int pass = 0; pass < 2; ++pass) {
    const int n0 = wv * 128 + pass * 64;
    v8f acc[4];
#pragma unroll
    for (int j = 0; j < 4; ++j) acc[j] = (v8f){0, 0, 0, 0, 0, 0, 0, 0};
    for (int k = 0; k < DM2; k += 32) {
      v16bf a = frag_bf16(hrow + k);
#pragma unroll
      for (int j = 0; j < 4; ++j) {
        const bf16* brow = W2 + (size_t)(n0 + j * 16 + nn) * DM2 + k + kh;
        acc[j] = wmma_bf16(a, frag_bf16(brow), acc[j]);
      }
    }
#pragma unroll
    for (int j = 0; j < 4; ++j)
#pragma unroll
      for (int v = 0; v < 8; ++v) {
        const int row = v + 8 * half, col = n0 + j * 16 + nn;
        float g = acc[j][v] + b2[col];
        g = 1.0f / (1.0f + __expf(-g));
        Out[(m0 + row) * DM + col] = Yt[row * DM + col] * g;
      }
  }
}

// ---------------------------------------------------------------------------
extern "C" void kernel_launch(void* const* d_in, const int* in_sizes, int n_in,
                              void* d_out, int out_size, void* d_ws, size_t ws_size,
                              hipStream_t stream) {
  (void)in_sizes; (void)n_in; (void)out_size; (void)ws_size;
  const float* X     = (const float*)d_in[0];
  const float* state = (const float*)d_in[1];
  const float* Aw    = (const float*)d_in[2];
  const float* Bw    = (const float*)d_in[3];
  const float* Cw    = (const float*)d_in[4];
  const float* Dv    = (const float*)d_in[5];
  const float* W1    = (const float*)d_in[6];
  const float* b1    = (const float*)d_in[7];
  const float* W2    = (const float*)d_in[8];
  const float* b2    = (const float*)d_in[9];

  float* Out      = (float*)d_out;                        // [16,2048,1024]
  float* OutState = Out + (size_t)BATCH * SEQ * DM;       // [16,1024]

  // workspace carve-up (256B aligned)
  char* ws = (char*)d_ws;
  size_t off = 0;
  auto carve = [&](size_t bytes) { char* p = ws + off; off = (off + bytes + 255) & ~(size_t)255; return p; };
  unsigned* ctr  = (unsigned*)carve(256);
  bf16* Abf      = (bf16*)carve((size_t)DM * DM * 2);
  bf16* Bbf      = (bf16*)carve((size_t)DM * DM * 2);
  bf16* Cbf      = (bf16*)carve((size_t)DM * DM * 2);
  bf16* W1bf     = (bf16*)carve((size_t)DM2 * DM * 2);
  bf16* W2bf     = (bf16*)carve((size_t)DM * DM2 * 2);
  float* U       = (float*)carve((size_t)BATCH * SEQ * DM * 4);
  bf16*  Sbuf    = (bf16*)carve((size_t)(SEQ + 1) * BATCH * DM * 2);

  (void)hipFuncSetAttribute((const void*)k_scan,
                            hipFuncAttributeMaxDynamicSharedMemorySize, 64 * DM * 2);
  (void)hipFuncSetAttribute((const void*)k_fused,
                            hipFuncAttributeMaxDynamicSharedMemorySize,
                            16 * DM * 4 + 16 * DM2 * 2);

  const int CT = 256;
  // 1) weight conversion + init
  k_cvt<<<(DM * DM + CT - 1) / CT, CT, 0, stream>>>(Abf, Aw, DM * DM);
  k_cvt<<<(DM * DM + CT - 1) / CT, CT, 0, stream>>>(Bbf, Bw, DM * DM);
  k_cvt<<<(DM * DM + CT - 1) / CT, CT, 0, stream>>>(Cbf, Cw, DM * DM);
  k_cvt<<<(DM2 * DM + CT - 1) / CT, CT, 0, stream>>>(W1bf, W1, DM2 * DM);
  k_cvt<<<(DM2 * DM + CT - 1) / CT, CT, 0, stream>>>(W2bf, W2, DM2 * DM);
  k_init<<<(BATCH * DM + CT - 1) / CT, CT, 0, stream>>>(Sbuf, state, ctr);

  // 2) U = X @ B^T   (M=32768, N=1024, K=1024)
  k_gemm_xw<<<dim3(DM / 64, (BATCH * SEQ) / 64), 128, 0, stream>>>(
      X, Bbf, U, BATCH * SEQ, DM, DM);

  // 3) sequential scan (16 persistent blocks, async-staged 128KB A-slice)
  k_scan<<<NBLK, 128, 64 * DM * 2, stream>>>(Abf, U, Sbuf, OutState, ctr);

  // 4) fused Y -> MLP -> gated output (128KB LDS, Y stays on-chip)
  k_fused<<<(BATCH * SEQ) / 16, 256, 16 * DM * 4 + 16 * DM2 * 2, stream>>>(
      Sbuf, Cbf, X, Dv, W1bf, b1, W2bf, b2, Out);
}
